// GraphDenoiser_57423712747675
// MI455X (gfx1250) — matrix operand order repaired and checked by reading, not compiled, not run on hardware
//
#include <hip/hip_runtime.h>
#include <hip/hip_bf16.h>

typedef __attribute__((ext_vector_type(16))) __bf16 v16bf;
typedef __attribute__((ext_vector_type(8)))  float  v8f;

#define GN    50000     // nodes   (= 3125 * 16)
#define GE    200000    // edges   (= 12500 * 16)
#define GBAT  1024
#define GD    256       // Kpad = 256 (8 chunks)
#define GIN0  392
#define GIN0P 416       // 392 padded to 32  (13 chunks)
#define GEINP 96        // 72  padded to 32  (3 chunks)
#define MAXKC 13        // max K chunks (416/32)

// ---------------------------------------------------------------------------
// Weight pack: W[K,N] (row-major f32) -> bf16 WMMA B-fragments, zero padded.
// P[(((tn*nkc + kc)*32 + lane)*16 + e)] = W[kc*32 + (lane>>4)*16 + e, tn*16 + (lane&15)]
// ---------------------------------------------------------------------------
__global__ void gdn_packB(const float* __restrict__ W, __bf16* __restrict__ P,
                          int K, int N, int nkc)
{
    long long idx = (long long)blockIdx.x * blockDim.x + threadIdx.x;
    int e  = (int)(idx & 15);
    int l  = (int)((idx >> 4) & 31);
    long long blk = idx >> 9;
    int kc = (int)(blk % nkc);
    int tn = (int)(blk / nkc);
    int ntn = (N + 15) >> 4;
    if (tn >= ntn) return;
    int k = kc * 32 + ((l >> 4) << 4) + e;
    int n = tn * 16 + (l & 15);
    float v = (k < K && n < N) ? W[(size_t)k * N + n] : 0.0f;
    P[idx] = (__bf16)v;
}

// ---------------------------------------------------------------------------
// WMMA GEMM:  Out[M,N] = act( bn( A[M,Kpad] @ W + bias ) )
// Block = 256 threads (8 waves) owns one 16-row M-tile and up to 16 N-tiles.
// A block (16 x Kpad) is cooperatively staged into LDS once, pre-converted to
// bf16 in the exact WMMA A-fragment layout; the K-loop is then pure
// ds_load (A) + global b128 (packed B) + v_wmma.
// M must be a multiple of 16 (true for all call sites).
// ---------------------------------------------------------------------------
__global__ __launch_bounds__(256) void gdn_wmma_gemm(
    const float* __restrict__ A, int lda,
    const __bf16* __restrict__ Bp,
    const float* __restrict__ bias,
    const float* __restrict__ gamma,
    const float* __restrict__ beta,
    float* __restrict__ Out, int ldo,
    int N, int nkc, int act)
{
    __shared__ __align__(32) __bf16 sA[MAXKC * 512];

    const int tid  = threadIdx.x;
    const int wave = tid >> 5;
    const int lane = tid & 31;
    const int ntn  = (N + 15) >> 4;
    const int m0   = blockIdx.x * 16;

    // ---- cooperative stage: A tile -> bf16 fragment layout in LDS ----
    // frag idx = kc*512 + l*16 + e  maps to  row = l&15,
    // k = kc*32 + ((l>>4)<<4) + e
    const int tot = nkc * 512;
    for (int idx = tid; idx < tot; idx += 256) {
        int e  = idx & 15;
        int l  = (idx >> 4) & 31;
        int kc = idx >> 9;
        int row = l & 15;
        int k   = kc * 32 + ((l >> 4) << 4) + e;
        sA[idx] = (__bf16)A[(size_t)(m0 + row) * lda + k];
    }
    __syncthreads();

    const int tn0 = (blockIdx.y * 8 + wave) * 2;
    if (tn0 >= ntn) return;                 // wave-uniform exit
    const int has2 = __builtin_amdgcn_readfirstlane((tn0 + 1 < ntn) ? 1 : 0);

    const int half = lane >> 4;
    const int l15  = lane & 15;

    const __bf16* __restrict__ Bt0 = Bp + ((size_t)tn0 * nkc * 32 + lane) * 16;
    const __bf16* __restrict__ Bt1 = Bt0 + (size_t)nkc * 32 * 16;
    const __bf16* sAl = sA + lane * 16;

    v8f acc0 = {}, acc1 = {};
    for (int kc = 0; kc < nkc; ++kc) {
        v16bf a  = *(const v16bf*)(sAl + kc * 512);
        v16bf b0 = *(const v16bf*)(Bt0 + (size_t)kc * 512);
        acc0 = __builtin_amdgcn_wmma_f32_16x16x32_bf16(
            false, a, false, b0, (short)0, acc0, false, false);
        if (has2) {
            v16bf b1 = *(const v16bf*)(Bt1 + (size_t)kc * 512);
            acc1 = __builtin_amdgcn_wmma_f32_16x16x32_bf16(
                false, a, false, b1, (short)0, acc1, false, false);
        }
    }

    const int rbase = m0 + 8 * half;
#pragma unroll
    for (int t = 0; t < 2; ++t) {
        if (t == 1 && !has2) break;
        int ncol = (tn0 + t) * 16 + l15;
        if (ncol >= N) continue;
        v8f acc = t ? acc1 : acc0;
        float bi = bias ? bias[ncol] : 0.0f;
        float g = 1.0f, bt = 0.0f;
        if (gamma) {                         // BN eval: gamma / sqrt(1 + 1e-5)
            g  = gamma[ncol] * 0.99999500003750f;
            bt = beta[ncol];
        }
#pragma unroll
        for (int r = 0; r < 8; ++r) {        // C layout: row = m0 + r + 8*half
            float v = acc[r] + bi;
            if (gamma) v = v * g + bt;
            if (act)   v = v * (1.0f / (1.0f + __expf(-v)));   // SiLU
            Out[(size_t)(rbase + r) * ldo + ncol] = v;
        }
    }
}

// ------------------------- element-wise helpers ----------------------------
__global__ void gdn_sinus(const int* __restrict__ t, float* __restrict__ T0, int n)
{
    int idx = blockIdx.x * blockDim.x + threadIdx.x;
    int i = idx >> 7, j = idx & 127;
    if (i >= n) return;
    float freq = __expf((float)j * -0.07252236512f);   // -ln(1e4)/127
    float ang  = (float)t[i] * freq;
    T0[(size_t)i * GD + j]       = __sinf(ang);
    T0[(size_t)i * GD + 128 + j] = __cosf(ang);
}

__global__ void gdn_cond(const float* __restrict__ cv, const float* __restrict__ Wc,
                         const float* __restrict__ bc, float* __restrict__ cg, int nb)
{
    int idx = blockIdx.x * blockDim.x + threadIdx.x;
    int b = idx >> 8, d = idx & 255;
    if (b >= nb) return;
    cg[(size_t)b * GD + d] = cv[b * 2] * Wc[d] + cv[b * 2 + 1] * Wc[GD + d] + bc[d];
}

__global__ void gdn_h0_embed(const int* __restrict__ xa, const int* __restrict__ nm,
                             const float* __restrict__ aemb, const float* __restrict__ memb,
                             float* __restrict__ h0, int n)
{
    int idx = blockIdx.x * blockDim.x + threadIdx.x;
    int i = idx / 136, j = idx % 136;
    if (i >= n) return;
    float v = (j < 128) ? aemb[xa[i] * 128 + j] : memb[nm[i] * 8 + (j - 128)];
    h0[(size_t)i * GIN0P + j] = v;
}

__global__ void gdn_add_cond(const int* __restrict__ batch, const float* __restrict__ cg,
                             float* __restrict__ h0, int n)
{
    int idx = blockIdx.x * blockDim.x + threadIdx.x;
    int i = idx >> 8, d = idx & 255;
    if (i >= n) return;
    h0[(size_t)i * GIN0P + 136 + d] += cg[(size_t)batch[i] * GD + d];
}

__global__ void gdn_ein(const int* __restrict__ et, const int* __restrict__ em,
                        const float* __restrict__ bemb, const float* __restrict__ memb,
                        float* __restrict__ ein, int ne)
{
    int idx = blockIdx.x * blockDim.x + threadIdx.x;
    int e = idx / 72, j = idx % 72;
    if (e >= ne) return;
    float v = (j < 64) ? bemb[et[e] * 64 + j] : memb[em[e] * 8 + (j - 64)];
    ein[(size_t)e * GEINP + j] = v;
}

__global__ void gdn_zero(float* __restrict__ p, long long n)
{
    long long i = (long long)blockIdx.x * blockDim.x + threadIdx.x;
    if (i < n) p[i] = 0.0f;
}

// msg = relu(h[src] + ef); agg[dst] += msg
__global__ void gdn_scatter(const int* __restrict__ src, const int* __restrict__ dst,
                            const float* __restrict__ h, int ldh,
                            const float* __restrict__ ef, int lde,
                            float* __restrict__ agg, int ldagg, int Dh, int ne)
{
    long long idx = (long long)blockIdx.x * blockDim.x + threadIdx.x;
    int e = (int)(idx / Dh), d = (int)(idx % Dh);
    if (e >= ne) return;
    float m = h[(size_t)src[e] * ldh + d] + ef[(size_t)e * lde + d];
    m = fmaxf(m, 0.0f);
    atomicAdd(&agg[(size_t)dst[e] * ldagg + d], m);
}

// z = (1+eps)*h + agg
__global__ void gdn_zcomb(const float* __restrict__ h, int ldh,
                          const float* __restrict__ agg, int ldagg,
                          const float* __restrict__ eps_p, int eps_i,
                          float* __restrict__ z, int ldz, int Dh, int n)
{
    long long idx = (long long)blockIdx.x * blockDim.x + threadIdx.x;
    int i = (int)(idx / Dh), d = (int)(idx % Dh);
    if (i >= n) return;
    float eps = eps_p[eps_i];
    z[(size_t)i * ldz + d] = (1.0f + eps) * h[(size_t)i * ldh + d] + agg[(size_t)i * ldagg + d];
}

__global__ void gdn_edgeh(const int* __restrict__ src, const int* __restrict__ dst,
                          const float* __restrict__ h, float* __restrict__ eh, int ne)
{
    long long idx = (long long)blockIdx.x * blockDim.x + threadIdx.x;
    int e = (int)(idx >> 8), d = (int)(idx & 255);
    if (e >= ne) return;
    eh[(size_t)e * GD + d] = h[(size_t)src[e] * GD + d] + h[(size_t)dst[e] * GD + d];
}

// ---------------------------------------------------------------------------
static inline int blks(long long n) { return (int)((n + 255) / 256); }

static inline void launch_pack(const float* W, __bf16* P, int K, int N, int Kpad,
                               hipStream_t s)
{
    int ntn = (N + 15) >> 4, nkc = Kpad >> 5;
    gdn_packB<<<blks((long long)ntn * nkc * 512), 256, 0, s>>>(W, P, K, N, nkc);
}

static inline void launch_gemm(const float* A, int lda, const __bf16* Bp,
                               const float* bias, const float* gamma, const float* beta,
                               float* Out, int ldo, int M, int N, int Kpad, int act,
                               hipStream_t s)
{
    int ntn = (N + 15) >> 4;
    dim3 grid(M / 16, (ntn + 15) / 16);     // 8 waves/block, 2 n-tiles/wave
    gdn_wmma_gemm<<<grid, 256, 0, s>>>(A, lda, Bp, bias, gamma, beta,
                                       Out, ldo, N, Kpad >> 5, act);
}

extern "C" void kernel_launch(void* const* d_in, const int* in_sizes, int n_in,
                              void* d_out, int out_size, void* d_ws, size_t ws_size,
                              hipStream_t stream)
{
    const int*   x_noisy  = (const int*)d_in[0];
    const int*   nmask    = (const int*)d_in[1];
    const int*   etype    = (const int*)d_in[2];
    const int*   emask    = (const int*)d_in[3];
    const int*   eidx     = (const int*)d_in[4];
    const int*   t_node   = (const int*)d_in[5];
    const float* cond_vec = (const float*)d_in[6];
    const int*   batch    = (const int*)d_in[7];
    const float* atom_emb = (const float*)d_in[8];
    const float* bond_emb = (const float*)d_in[9];
    const float* nm_emb   = (const float*)d_in[10];
    const float* em_emb   = (const float*)d_in[11];
    const float* t_W1     = (const float*)d_in[12];
    const float* t_b1     = (const float*)d_in[13];
    const float* t_W2     = (const float*)d_in[14];
    const float* t_b2     = (const float*)d_in[15];
    const float* cond_W   = (const float*)d_in[16];
    const float* cond_b   = (const float*)d_in[17];
    const float* edge_W1  = (const float*)d_in[18];
    const float* edge_b1  = (const float*)d_in[19];
    const float* edge_W2  = (const float*)d_in[20];
    const float* edge_b2  = (const float*)d_in[21];
    const float* e0_W     = (const float*)d_in[22];
    const float* c0_W1    = (const float*)d_in[23];
    const float* c0_b1    = (const float*)d_in[24];
    const float* c0_W2    = (const float*)d_in[25];
    const float* c0_b2    = (const float*)d_in[26];
    const float* c0_eps   = (const float*)d_in[27];
    const float* cs_W1    = (const float*)d_in[28];
    const float* cs_b1    = (const float*)d_in[29];
    const float* cs_W2    = (const float*)d_in[30];
    const float* cs_b2    = (const float*)d_in[31];
    const float* cs_eps   = (const float*)d_in[32];
    const float* bn_g     = (const float*)d_in[33];
    const float* bn_b     = (const float*)d_in[34];
    const float* atom_W   = (const float*)d_in[35];
    const float* atom_b   = (const float*)d_in[36];
    const float* bond_W   = (const float*)d_in[37];
    const float* bond_b   = (const float*)d_in[38];
    float* out = (float*)d_out;

    const int* src = eidx;
    const int* dst = eidx + GE;

    // --- workspace layout (floats) ---
    float* ws   = (float*)d_ws;
    float* bufA = ws;                          // E*392 : e_in(96) -> e_proj(392) -> edge_h(256)
    float* bufB = bufA + (size_t)GE * GIN0;    // E*256 : e_feat (persistent)
    float* bufC = bufB + (size_t)GE * GD;      // E*256 : e_tmp -> h ping/pong
    float* bufD = bufC + (size_t)GE * GD;      // N*416 : h0 (padded) -> conv hidden (256)
    float* bufE = bufD + (size_t)GN * GIN0P;   // N*392 : T0 -> agg
    float* bufF = bufE + (size_t)GN * GIN0;    // N*416 : T1 (256) -> z (padded)
    float* bufG = bufF + (size_t)GN * GIN0P;   // B*256 : cond per graph
    float* hP   = bufC;
    float* hQ   = bufC + (size_t)GN * GD;

    // --- packed bf16 weights (after float buffers) ---
    __bf16* pk = (__bf16*)(bufG + (size_t)GBAT * GD);
    const int F256x256 = 16 * 8 * 512;                     // 65536 frags
    __bf16* p_tW1  = pk;                 pk += F256x256;
    __bf16* p_tW2  = pk;                 pk += F256x256;
    __bf16* p_eW1  = pk;                 pk += 16 * 3 * 512;     // K=72->96
    __bf16* p_eW2  = pk;                 pk += F256x256;
    __bf16* p_e0W  = pk;                 pk += 25 * 8 * 512;     // N=392
    __bf16* p_c0W1 = pk;                 pk += 16 * 13 * 512;    // K=392->416
    __bf16* p_c0W2 = pk;                 pk += F256x256;
    __bf16* p_csW1 = pk;                 pk += 5 * F256x256;
    __bf16* p_csW2 = pk;                 pk += 5 * F256x256;
    __bf16* p_atW  = pk;                 pk += 1 * 8 * 512;      // N=16
    __bf16* p_bdW  = pk;                 /* last */

    // ---- pack all weights (tiny, reused by thousands of M-tiles) ----
    launch_pack(t_W1,   p_tW1,  GD,  GD,   GD,    stream);
    launch_pack(t_W2,   p_tW2,  GD,  GD,   GD,    stream);
    launch_pack(edge_W1,p_eW1,  72,  GD,   GEINP, stream);
    launch_pack(edge_W2,p_eW2,  GD,  GD,   GD,    stream);
    launch_pack(e0_W,   p_e0W,  GD,  GIN0, GD,    stream);
    launch_pack(c0_W1,  p_c0W1, GIN0,GD,   GIN0P, stream);
    launch_pack(c0_W2,  p_c0W2, GD,  GD,   GD,    stream);
    for (int i = 0; i < 5; ++i) {
        launch_pack(cs_W1 + (size_t)i * GD * GD, p_csW1 + (size_t)i * F256x256, GD, GD, GD, stream);
        launch_pack(cs_W2 + (size_t)i * GD * GD, p_csW2 + (size_t)i * F256x256, GD, GD, GD, stream);
    }
    launch_pack(atom_W, p_atW,  GD, 16, GD, stream);
    launch_pack(bond_W, p_bdW,  GD, 5,  GD, stream);

    // ---- node features: h0 [N, 392 (pad 416)] ----
    gdn_zero<<<blks((long long)GN * GIN0P), 256, 0, stream>>>(bufD, (long long)GN * GIN0P);
    gdn_sinus<<<blks((long long)GN * 128), 256, 0, stream>>>(t_node, bufE, GN);
    launch_gemm(bufE, GD, p_tW1, t_b1, nullptr, nullptr, bufF, GD, GN, GD, GD, 1, stream);
    launch_gemm(bufF, GD, p_tW2, t_b2, nullptr, nullptr, bufD + 136, GIN0P, GN, GD, GD, 0, stream);
    gdn_cond<<<blks((long long)GBAT * GD), 256, 0, stream>>>(cond_vec, cond_W, cond_b, bufG, GBAT);
    gdn_h0_embed<<<blks((long long)GN * 136), 256, 0, stream>>>(x_noisy, nmask, atom_emb, nm_emb, bufD, GN);
    gdn_add_cond<<<blks((long long)GN * GD), 256, 0, stream>>>(batch, bufG, bufD, GN);

    // ---- edge features: e_feat [E,256] ----
    gdn_zero<<<blks((long long)GE * GEINP), 256, 0, stream>>>(bufA, (long long)GE * GEINP);
    gdn_ein<<<blks((long long)GE * 72), 256, 0, stream>>>(etype, emask, bond_emb, em_emb, bufA, GE);
    launch_gemm(bufA, GEINP, p_eW1, edge_b1, nullptr, nullptr, bufC, GD, GE, GD, GEINP, 1, stream);
    launch_gemm(bufC, GD, p_eW2, edge_b2, nullptr, nullptr, bufB, GD, GE, GD, GD, 0, stream);

    // ---- layer 0 ----
    launch_gemm(bufB, GD, p_e0W, nullptr, nullptr, nullptr, bufA, GIN0, GE, GIN0, GD, 0, stream);
    gdn_zero<<<blks((long long)GN * GIN0), 256, 0, stream>>>(bufE, (long long)GN * GIN0);
    gdn_scatter<<<blks((long long)GE * GIN0), 256, 0, stream>>>(src, dst, bufD, GIN0P, bufA, GIN0, bufE, GIN0, GIN0, GE);
    gdn_zero<<<blks((long long)GN * GIN0P), 256, 0, stream>>>(bufF, (long long)GN * GIN0P);
    gdn_zcomb<<<blks((long long)GN * GIN0), 256, 0, stream>>>(bufD, GIN0P, bufE, GIN0, c0_eps, 0, bufF, GIN0P, GIN0, GN);
    launch_gemm(bufF, GIN0P, p_c0W1, c0_b1, nullptr, nullptr, bufD, GD, GN, GD, GIN0P, 1, stream);
    launch_gemm(bufD, GD, p_c0W2, c0_b2, bn_g, bn_b, hP, GD, GN, GD, GD, 1, stream);

    // ---- layers 1..5 ----
    float* hcur = hP; float* hnxt = hQ;
    for (int i = 0; i < 5; ++i) {
        gdn_zero<<<blks((long long)GN * GD), 256, 0, stream>>>(bufE, (long long)GN * GD);
        gdn_scatter<<<blks((long long)GE * GD), 256, 0, stream>>>(src, dst, hcur, GD, bufB, GD, bufE, GD, GD, GE);
        gdn_zcomb<<<blks((long long)GN * GD), 256, 0, stream>>>(hcur, GD, bufE, GD, cs_eps, i, bufF, GIN0P, GD, GN);
        launch_gemm(bufF, GIN0P, p_csW1 + (size_t)i * F256x256, cs_b1 + i * GD,
                    nullptr, nullptr, bufD, GD, GN, GD, GD, 1, stream);
        launch_gemm(bufD, GD, p_csW2 + (size_t)i * F256x256, cs_b2 + i * GD,
                    bn_g + (i + 1) * GD, bn_b + (i + 1) * GD, hnxt, GD, GN, GD, GD, 1, stream);
        float* t = hcur; hcur = hnxt; hnxt = t;
    }

    // ---- heads ----
    launch_gemm(hcur, GD, p_atW, atom_b, nullptr, nullptr, out, 16, GN, 16, GD, 0, stream);
    gdn_edgeh<<<blks((long long)GE * GD), 256, 0, stream>>>(src, dst, hcur, bufA, GE);
    launch_gemm(bufA, GD, p_bdW, bond_b, nullptr, nullptr, out + (size_t)GN * 16, 5, GE, 5, GD, 0, stream);
}
